// RNN_13821204759023
// MI455X (gfx1250) — matrix operand order repaired and checked by reading, not compiled, not run on hardware
//
#include <hip/hip_runtime.h>

#define B_  4096
#define T_  128
#define D_  64
#define U_  128
#define G_  512   // 4*U
#define C_  10

typedef __attribute__((ext_vector_type(16))) __bf16 v16bf;
typedef __attribute__((ext_vector_type(8)))  __bf16 v8bf;
typedef __attribute__((ext_vector_type(4)))  __bf16 v4bf;
typedef __attribute__((ext_vector_type(8)))  float  v8f;
typedef __attribute__((ext_vector_type(4)))  float  v4f;
typedef __attribute__((ext_vector_type(4)))  unsigned int u32x4;
typedef __attribute__((ext_vector_type(8)))  int    i32x8;
typedef __attribute__((ext_vector_type(4)))  int    i32x4;

#if defined(__has_builtin)
#if __has_builtin(__builtin_amdgcn_tensor_load_to_lds)
#define HAVE_TDM 1
#else
#define HAVE_TDM 0
#endif
#else
#define HAVE_TDM 0
#endif

#if __has_include(<hip/amd_detail/amd_gfx1250_TDM.h>)
#define TDM_6ARG 1
#else
#define TDM_6ARG 0
#endif

__device__ __forceinline__ void wait_tensor0() {
#if __has_builtin(__builtin_amdgcn_s_wait_tensorcnt)
  __builtin_amdgcn_s_wait_tensorcnt(0);
#else
  asm volatile("s_wait_tensorcnt 0x0" ::: "memory");
#endif
}

#if HAVE_TDM
// 2D TDM descriptor: tile = 64 fp32 elems x 16 rows, row stride = T*D elements.
__device__ __forceinline__ void tdm_load_x_tile(const float* gptr, unsigned lds_off) {
  unsigned long long ga = (unsigned long long)gptr;
  u32x4 g0 = { 1u,                                    // count=1, user descriptor
               lds_off,                               // LDS byte address
               (unsigned)(ga & 0xFFFFFFFFu),          // global_addr[31:0]
               (unsigned)((ga >> 32) & 0x1FFFFFFu) | 0x80000000u }; // addr[56:32] | type=2
  i32x8 g1 = { (int)(2u << 16),                      // data_size = 4B
               (int)(64u << 16),                     // tensor_dim0 = 64 (low16)
               (int)(16u << 16),                     // dim0 hi=0 | tensor_dim1 = 16
               (int)(64u << 16),                     // dim1 hi=0 | tile_dim0 = 64
               16,                                   // tile_dim1 = 16, tile_dim2 = 0
               (int)(T_ * D_),                       // tensor_dim0_stride = 8192
               0, 0 };
  i32x4 z4 = { 0, 0, 0, 0 };
#if TDM_6ARG
  i32x8 z8 = { 0, 0, 0, 0, 0, 0, 0, 0 };
  __builtin_amdgcn_tensor_load_to_lds(g0, g1, z4, z4, z8, 0);
#else
  __builtin_amdgcn_tensor_load_to_lds(g0, g1, z4, z4, 0);
#endif
}
#endif

// Hardware reciprocal (v_rcp_f32) based activations — no software division.
__device__ __forceinline__ float fast_rcp(float x) { return __builtin_amdgcn_rcpf(x); }
__device__ __forceinline__ float sigmoid_f(float z) {
  return fast_rcp(1.0f + __expf(-z));
}
__device__ __forceinline__ float tanh_f(float z) {
  // tanh(z) = 2*sigmoid(2z) - 1; exp overflow/underflow saturate correctly via rcp.
  return __builtin_fmaf(2.0f, fast_rcp(1.0f + __expf(-2.0f * z)), -1.0f);
}

__global__ __launch_bounds__(256, 1)
void lstm_fused_kernel(const float* __restrict__ x,
                       const float* __restrict__ Wk,
                       const float* __restrict__ R,
                       const float* __restrict__ bias,
                       const float* __restrict__ Wd,
                       const float* __restrict__ bd,
                       float* __restrict__ out)
{
  __shared__ __bf16 sRT[G_ * U_];        // transposed recurrent kernel (131072 B)
  __shared__ float  sRaw[2][16 * D_];    // TDM fp32 staging            (8192 B)
  __shared__ __bf16 sXbf[2][16 * D_];    // converted x tiles           (4096 B)
  __shared__ __bf16 sH[2][16 * U_];      // hidden state                (8192 B)

  const int tid  = threadIdx.x;
  const int lane = tid & 31;
  const int w    = tid >> 5;             // wave 0..7
  const int b0   = blockIdx.x * 16;      // batch tile origin

  // ---------------- setup: transpose R into LDS (fp32 -> bf16) ----------------
  for (int idx = tid; idx < U_ * G_; idx += 256) {
    int k = idx >> 9;                    // row of R
    int n = idx & (G_ - 1);              // col of R
    sRT[n * U_ + k] = (__bf16)R[idx];
  }
  for (int idx = tid; idx < 16 * U_; idx += 256) sH[0][idx] = (__bf16)0.0f;

  const int  nlo  = lane & 15;
  const bool hi   = lane >= 16;
  const int  ucol = w * 16 + nlo;        // this wave's u-column

  const float b_i = bias[ucol];
  const float b_f = bias[U_ + ucol];
  const float b_g = bias[2 * U_ + ucol];
  const float b_o = bias[3 * U_ + ucol];

  // ---------------- Wk B-tiles resident in registers (bf16) ----------------
  v16bf wkb[4][2];
#pragma unroll
  for (int j = 0; j < 4; ++j) {
    const int n = j * U_ + ucol;
#pragma unroll
    for (int kk = 0; kk < 2; ++kk) {
      const int kb = kk * 32 + (hi ? 16 : 0);
#pragma unroll
      for (int v = 0; v < 8; ++v) {
        wkb[j][kk][2 * v]     = (__bf16)Wk[(kb + 2 * v) * G_ + n];
        wkb[j][kk][2 * v + 1] = (__bf16)Wk[(kb + 2 * v + 1) * G_ + n];
      }
    }
  }

  v8f c = {0.f, 0.f, 0.f, 0.f, 0.f, 0.f, 0.f, 0.f};

  // cooperative fp32->bf16 tile conversion: 4 elements per thread
  auto convert_tile = [&](int buf) {
    const int i0 = tid * 4;
    v4f f = *(const v4f*)&sRaw[buf][i0];
    v4bf b;
    b[0] = (__bf16)f[0]; b[1] = (__bf16)f[1];
    b[2] = (__bf16)f[2]; b[3] = (__bf16)f[3];
    *(v4bf*)&sXbf[buf][i0] = b;
  };

  // ---------------- prologue: stage x tile for t=0 (and launch t=1) ----------
#if HAVE_TDM
  if (w == 0) {
    tdm_load_x_tile(x + (unsigned long long)b0 * (T_ * D_),
                    (unsigned)(unsigned long long)&sRaw[0][0]);
    wait_tensor0();
  }
  __syncthreads();
  convert_tile(0);
  if (w == 0)
    tdm_load_x_tile(x + (unsigned long long)b0 * (T_ * D_) + D_,
                    (unsigned)(unsigned long long)&sRaw[1][0]);
#else
  {
    const int i0 = tid * 4;
    const int r = i0 >> 6, d = i0 & 63;
    v4f f = *(const v4f*)&x[(unsigned long long)(b0 + r) * (T_ * D_) + d];
    v4bf b;
    b[0] = (__bf16)f[0]; b[1] = (__bf16)f[1];
    b[2] = (__bf16)f[2]; b[3] = (__bf16)f[3];
    *(v4bf*)&sXbf[0][i0] = b;
  }
#endif
  __syncthreads();

  // ---------------- timestep loop ----------------
#pragma unroll 1
  for (int t = 0; t < T_; ++t) {
    const int cur = t & 1, nxt = cur ^ 1;
    const int m = nlo;                   // A-matrix row = lane&15

    // ---- A tiles (pure ds_load_b128, 16x32 wave32 layout) ----
    v16bf xa[2];
#pragma unroll
    for (int kk = 0; kk < 2; ++kk) {
      const int c0 = kk * 32 + (hi ? 16 : 0);
      union { v16bf v; v8bf p[2]; } u;
      u.p[0] = *(const v8bf*)&sXbf[cur][m * D_ + c0];
      u.p[1] = *(const v8bf*)&sXbf[cur][m * D_ + c0 + 8];
      xa[kk] = u.v;
    }
    v16bf ha[4];
#pragma unroll
    for (int kk = 0; kk < 4; ++kk) {
      const int c0 = kk * 32 + (hi ? 16 : 0);
      union { v16bf v; v8bf p[2]; } u;
      u.p[0] = *(const v8bf*)&sH[cur][m * U_ + c0];
      u.p[1] = *(const v8bf*)&sH[cur][m * U_ + c0 + 8];
      ha[kk] = u.v;
    }

    // ---- z = x_t @ Wk + h @ R for this wave's 4 gate tiles ----
    v8f acc[4];
#pragma unroll
    for (int j = 0; j < 4; ++j) {
      v8f a = {0.f, 0.f, 0.f, 0.f, 0.f, 0.f, 0.f, 0.f};
#pragma unroll
      for (int kk = 0; kk < 2; ++kk)
        a = __builtin_amdgcn_wmma_f32_16x16x32_bf16(false, xa[kk], false, wkb[j][kk],
                                                    (short)0, a, false, false);
      const int n = j * U_ + ucol;
#pragma unroll
      for (int kk = 0; kk < 4; ++kk) {
        const int kb = kk * 32 + (hi ? 16 : 0);
        union { v16bf v; v8bf p[2]; } u;
        u.p[0] = *(const v8bf*)&sRT[n * U_ + kb];
        u.p[1] = *(const v8bf*)&sRT[n * U_ + kb + 8];
        a = __builtin_amdgcn_wmma_f32_16x16x32_bf16(false, ha[kk], false, u.v,
                                                    (short)0, a, false, false);
      }
      acc[j] = a;
    }

    // ---- gates + state update (C/D layout: VGPR r -> M=r / r+8) ----
#pragma unroll
    for (int r = 0; r < 8; ++r) {
      float ig = sigmoid_f(acc[0][r] + b_i);
      float fg = sigmoid_f(acc[1][r] + b_f);
      float gt = tanh_f(acc[2][r] + b_g);
      float og = sigmoid_f(acc[3][r] + b_o);
      float cn = fg * c[r] + ig * gt;
      c[r] = cn;
      float hn = og * tanh_f(cn);
      sH[nxt][(r + (hi ? 8 : 0)) * U_ + ucol] = (__bf16)hn;
    }

    // ---- stage next x tile ----
#if HAVE_TDM
    if (w == 0 && t + 1 < T_) wait_tensor0();    // raw[nxt] landed
    __syncthreads();                             // sH[nxt] + raw[nxt] visible
    if (t + 1 < T_) convert_tile(nxt);
    if (w == 0 && t + 2 < T_)
      tdm_load_x_tile(x + (unsigned long long)b0 * (T_ * D_) + (unsigned)(t + 2) * D_,
                      (unsigned)(unsigned long long)&sRaw[cur][0]);
    __syncthreads();                             // sXbf[nxt] visible
#else
    __syncthreads();
    if (t + 1 < T_) {
      const int i0 = tid * 4;
      const int r = i0 >> 6, d = i0 & 63;
      v4f f = *(const v4f*)&x[(unsigned long long)(b0 + r) * (T_ * D_) + (unsigned)(t + 1) * D_ + d];
      v4bf b;
      b[0] = (__bf16)f[0]; b[1] = (__bf16)f[1];
      b[2] = (__bf16)f[2]; b[3] = (__bf16)f[3];
      *(v4bf*)&sXbf[nxt][i0] = b;
    }
    __syncthreads();
#endif
  }

  // ---------------- dense head + softmax (final h is in sH[0]) ----------------
  if (tid < 16) {
    const int b = b0 + tid;
    float logit[C_];
    float mx = -3.4e38f;
#pragma unroll
    for (int cc = 0; cc < C_; ++cc) {
      float s = bd[cc];
#pragma unroll 4
      for (int k = 0; k < U_; ++k)
        s += (float)sH[0][tid * U_ + k] * Wd[k * C_ + cc];
      logit[cc] = s;
      mx = fmaxf(mx, s);
    }
    float den = 0.f;
#pragma unroll
    for (int cc = 0; cc < C_; ++cc) { logit[cc] = __expf(logit[cc] - mx); den += logit[cc]; }
    float rden = fast_rcp(den);
#pragma unroll
    for (int cc = 0; cc < C_; ++cc) out[b * C_ + cc] = logit[cc] * rden;
  }
}

extern "C" void kernel_launch(void* const* d_in, const int* in_sizes, int n_in,
                              void* d_out, int out_size, void* d_ws, size_t ws_size,
                              hipStream_t stream) {
  const float* x    = (const float*)d_in[0];
  const float* Wk   = (const float*)d_in[1];
  const float* R    = (const float*)d_in[2];
  const float* bias = (const float*)d_in[3];
  const float* Wd   = (const float*)d_in[4];
  const float* bd   = (const float*)d_in[5];
  float* out = (float*)d_out;

  dim3 grid(B_ / 16);    // 256 workgroups, one 16-row batch tile each
  dim3 block(256);       // 8 waves (wave32)
  hipLaunchKernelGGL(lstm_fused_kernel, grid, block, 0, stream,
                     x, Wk, R, bias, Wd, bd, out);
}
